// Gate_34746285425193
// MI455X (gfx1250) — compile-verified, matrix operand order checked by reading, not run on hardware
//
#include <hip/hip_runtime.h>
#include <hip/hip_bf16.h>

// ---------------------------------------------------------------------------
// MoE conv-gate for MI455X (gfx1250, wave32, WMMA + async global->LDS)
//   conv3x3(256->64) as 9 shifted GEMMs, bf16 WMMA 16x16x32, f32 accum,
//   async-copy weight staging, fused sigmoid + top-8 + softmax + bincount.
// ---------------------------------------------------------------------------

typedef __attribute__((ext_vector_type(16))) __bf16 v16bf;
typedef __attribute__((ext_vector_type(8)))  float  v8f;

namespace {
constexpr int Bn   = 8;
constexpr int CIN  = 256;
constexpr int HH   = 128;
constexpr int WW   = 128;
constexpr int NE   = 64;      // experts
constexpr int TOPK = 8;
constexpr float WRAPV = 1e8f;
constexpr float URATE = 0.001f;

// per-channel-chunk weight block: [9 tap][64 e][32 c] bf16 = 36,864 B
constexpr int WBLK_ELEMS = 9 * NE * 32;                     // 18,432 bf16
// d_ws layout
constexpr size_t WS_WB_BYTES  = (size_t)(CIN / 32) * WBLK_ELEMS * 2;  // 294,912 B
constexpr size_t WS_CNT_OFF   = WS_WB_BYTES;                // 64 ints

// dynamic LDS layout (bytes)
constexpr int XS_OFF   = 0;          // bf16 x stage: [3 rows][130 w][32 c] = 24,960 B
constexpr int WL_OFF   = 24960;      // bf16 w stage: [9 tap][64 e][32 c]  = 36,864 B
constexpr int SC_OFF   = 0;          // f32 scores [128 pix][64 e] = 32,768 B (aliases stages)
constexpr int BIAS_OFF = 61824;      // 64 f32
constexpr int CNT_OFF  = 62080;      // 64 i32
constexpr int SMEM_BYTES = 62336;
}

__device__ __forceinline__ unsigned short f2bf(float f) {
  // round-to-nearest-even fp32 -> bf16
  unsigned u = __float_as_uint(f);
  unsigned r = (u + 0x7FFFu + ((u >> 16) & 1u)) >> 16;
  return (unsigned short)r;
}

// ---------------------------------------------------------------------------
// Kernel 0: repack conv weights OIHW fp32 -> bf16, chunked LDS-ready layout:
//   wb[ ((cb*9 + tap)*64 + e)*32 + c_local ],  cb = c>>5, c_local = c&31
// Also zeroes the global count scratch (deterministic per launch).
// ---------------------------------------------------------------------------
__global__ __launch_bounds__(256) void prep_kernel(
    const float* __restrict__ wg, unsigned short* __restrict__ wb,
    int* __restrict__ cnt)
{
  int i = blockIdx.x * 256 + threadIdx.x;
  if (i < NE * CIN * 9) {
    int e  = i / (CIN * 9);
    int ct = i - e * (CIN * 9);          // c*9 + tap
    int c  = ct / 9;
    int t  = ct - c * 9;
    int cb = c >> 5, cl = c & 31;
    wb[(size_t)((cb * 9 + t) * NE + e) * 32 + cl] = f2bf(wg[i]);
  }
  if (i < NE) cnt[i] = 0;
}

// ---------------------------------------------------------------------------
// Kernel 1: one block per (b, h) row. 256 thr = 8 waves. Wave w owns pixels
// [w*16, w*16+16) x all 64 experts (4 accumulator tiles).
// ---------------------------------------------------------------------------
__global__ __launch_bounds__(256) void moe_gate_conv(
    const float* __restrict__ x, const unsigned short* __restrict__ wb,
    const float* __restrict__ bias, float* __restrict__ out,
    int* __restrict__ gcnt)
{
  extern __shared__ char smem[];
  unsigned short* xs  = (unsigned short*)(smem + XS_OFF);   // [kh][w+1][c]
  unsigned short* wl  = (unsigned short*)(smem + WL_OFF);   // [tap][e][c]
  unsigned*       xsu = (unsigned*)(smem + XS_OFF);
  unsigned*       wlu = (unsigned*)(smem + WL_OFF);
  float*          sc  = (float*)(smem + SC_OFF);            // [pix][e]
  float*          bsh = (float*)(smem + BIAS_OFF);
  int*            csh = (int*)(smem + CNT_OFF);

  const int tid  = threadIdx.x;
  const int lane = tid & 31;
  const int wave = tid >> 5;
  const int bid  = blockIdx.x;
  const int b = bid >> 7;
  const int h = bid & 127;

  if (tid < NE) { bsh[tid] = bias[tid]; csh[tid] = 0; }

  const int m   = lane & 15;    // row within 16x16 tile
  const int hi  = lane >> 4;    // K-half select per ISA A/B layout
  const int pix = wave * 16 + m;

  // LDS byte address of the weight stage (generic LDS addr keeps offset in [31:0])
  const unsigned wl_lds_base = (unsigned)(unsigned long long)(void*)wl;

  v8f acc[4] = {};

  for (int cb = 0; cb < CIN / 32; ++cb) {
    const int cbase = cb * 32;

    // ---- async global->LDS copy of this chunk's weight block (36,864 B) ----
    // Contiguous bf16 block, prepacked in LDS layout by prep_kernel.
    {
      const unsigned short* gsrc = wb + (size_t)cb * WBLK_ELEMS;
      #pragma unroll
      for (int k = 0; k < 9; ++k) {                 // 2304 x b128 across 256 thr
        int idx = tid + k * 256;
        unsigned ldsa = wl_lds_base + (unsigned)(idx * 16);
        int voff = idx * 16;
        asm volatile("global_load_async_to_lds_b128 %0, %1, %2"
                     :: "v"(ldsa), "v"(voff), "s"(gsrc)
                     : "memory");
      }
    }

    // ---- stage x rows h-1,h,h+1 for 32 channels (coalesced over w),
    //      overlapped with the async weight copy ----
    #pragma unroll 1
    for (int i = tid; i < 3 * 32 * 128; i += 256) {
      int w  = i & 127;
      int c  = (i >> 7) & 31;
      int kh = i >> 12;
      int r  = h + kh - 1;
      float v = 0.f;
      if ((unsigned)r < (unsigned)HH)
        v = x[(((size_t)b * CIN + (cbase + c)) * HH + r) * WW + w];
      xs[(kh * 130 + w + 1) * 32 + c] = f2bf(v);
    }
    if (tid < 192) {  // zero halo columns w=-1, w=128
      int c = tid & 31, side = (tid >> 5) & 1, kh = tid >> 6;
      xs[(kh * 130 + (side ? 129 : 0)) * 32 + c] = 0;
    }

    asm volatile("s_wait_asynccnt 0x0" ::: "memory");
    __syncthreads();

    // ---- 9 taps x 4 expert tiles of WMMA (b128 fragment loads) ----
    #pragma unroll
    for (int kh = 0; kh < 3; ++kh) {
      #pragma unroll
      for (int kw = 0; kw < 3; ++kw) {
        const int tap = kh * 3 + kw;
        // A fragment: pixels x 32 channels, shifted by (kh,kw)
        union { v16bf v; uint4 q[2]; } A;
        const int arow = (kh * 130 + pix + kw) * 16;  // uint index base
        A.q[0] = *(const uint4*)(xsu + arow + hi * 4);
        A.q[1] = *(const uint4*)(xsu + arow + 8 + hi * 4);
        #pragma unroll
        for (int n = 0; n < 4; ++n) {
          union { v16bf v; uint4 q[2]; } Bf;
          const int brow = (tap * NE + n * 16 + m) * 16;
          Bf.q[0] = *(const uint4*)(wlu + brow + hi * 4);
          Bf.q[1] = *(const uint4*)(wlu + brow + 8 + hi * 4);
          acc[n] = __builtin_amdgcn_wmma_f32_16x16x32_bf16(
              false, A.v, false, Bf.v, (short)0, acc[n], false, false);
        }
      }
    }
    __syncthreads();   // stages (and, after last iter, accum reads) done
  }

  // ---- sigmoid -> scores LDS (C layout: VGPR r -> M=r+8*hi, N=lane&15) ----
  #pragma unroll
  for (int n = 0; n < 4; ++n) {
    #pragma unroll
    for (int r = 0; r < 8; ++r) {
      int p = wave * 16 + r + 8 * hi;
      int e = n * 16 + m;
      float z = acc[n][r];
      sc[p * NE + e] = 1.f / (1.f + __expf(-z));
    }
  }
  __syncthreads();

  // ---- per-pixel top-8 of 64 (biased), softmax over unbiased, outputs ----
  if (tid < 128) {
    const float* row = sc + tid * NE;
    unsigned long long chosen = 0ull;
    float selv[TOPK];
    int   seli[TOPK];
    #pragma unroll 1
    for (int k = 0; k < TOPK; ++k) {
      float best = -3.4e38f; int bi = 0;
      #pragma unroll 1
      for (int e = 0; e < NE; ++e) {
        if ((chosen >> e) & 1ull) continue;
        float v = row[e] + bsh[e];
        if (v > best) { best = v; bi = e; }   // strict '>' == jax tie-break
      }
      chosen |= 1ull << bi;
      seli[k] = bi;
      selv[k] = row[bi];                      // unbiased sigmoid score
    }
    float mx = selv[0];
    #pragma unroll
    for (int k = 1; k < TOPK; ++k) mx = fmaxf(mx, selv[k]);
    float s = 0.f;
    #pragma unroll
    for (int k = 0; k < TOPK; ++k) { selv[k] = __expf(selv[k] - mx); s += selv[k]; }
    float inv = 1.f / s;  // ROUTE_SCALE == 1
    const size_t HW = (size_t)HH * WW;
    size_t obase = (size_t)b * TOPK * HW + (size_t)h * WW + tid;
    #pragma unroll
    for (int k = 0; k < TOPK; ++k) {
      size_t o = obase + (size_t)k * HW;
      out[o] = selv[k] * inv;                            // weights (B,K,H,W)
      out[(size_t)Bn * TOPK * HW + o] = (float)seli[k];  // indices
      atomicAdd(&csh[seli[k]], 1);
    }
  }
  __syncthreads();
  if (tid < NE && csh[tid] != 0) atomicAdd(&gcnt[tid], csh[tid]);
}

// ---------------------------------------------------------------------------
// Kernel 2: counts + wrap + mean + bias sign-update
// ---------------------------------------------------------------------------
__global__ __launch_bounds__(64) void finalize_kernel(
    const float* __restrict__ bias, const float* __restrict__ hist,
    const int* __restrict__ cnt, float* __restrict__ out)
{
  __shared__ float s[NE];
  __shared__ float meansh;
  __shared__ int allf;
  int e = threadIdx.x;
  float c = hist[e] + (float)cnt[e];
  if (e == 0) allf = 1;
  __syncthreads();
  if (!(c > WRAPV)) atomicAnd(&allf, 0);
  __syncthreads();
  float cw = allf ? fmodf(c, WRAPV) : c;
  s[e] = cw;
  __syncthreads();
  if (e == 0) {
    float t = 0.f;
    for (int i = 0; i < NE; ++i) t += s[i];
    meansh = t / (float)NE;
  }
  __syncthreads();
  const size_t OUTC = (size_t)Bn * TOPK * HH * WW * 2;  // 2,097,152
  out[OUTC + e] = cw;
  float diff = meansh - cw;
  float sg = (diff > 0.f) ? 1.f : ((diff < 0.f) ? -1.f : 0.f);
  out[OUTC + NE + e] = bias[e] + URATE * sg;
}

// ---------------------------------------------------------------------------
extern "C" void kernel_launch(void* const* d_in, const int* in_sizes, int n_in,
                              void* d_out, int out_size, void* d_ws, size_t ws_size,
                              hipStream_t stream) {
  const float* x    = (const float*)d_in[0];
  const float* wg   = (const float*)d_in[1];
  const float* bias = (const float*)d_in[2];
  const float* hist = (const float*)d_in[3];
  float* out = (float*)d_out;

  unsigned short* wbp = (unsigned short*)d_ws;
  int* cnt = (int*)((char*)d_ws + WS_CNT_OFF);

  // repack weights to bf16 (LDS-ready chunk layout) + zero count scratch
  int prep_elems = NE * CIN * 9;
  prep_kernel<<<(prep_elems + 255) / 256, 256, 0, stream>>>(wg, wbp, cnt);

  // one block per (b, h) row: 8*128 = 1024 blocks
  moe_gate_conv<<<Bn * HH, 256, SMEM_BYTES, stream>>>(x, wbp, bias, out, cnt);

  finalize_kernel<<<1, NE, 0, stream>>>(bias, hist, cnt, out);
}
